// ALPHATorch_64656437673993
// MI455X (gfx1250) — matrix operand order repaired and checked by reading, not compiled, not run on hardware
//
#include <hip/hip_runtime.h>
#include <hip/hip_bf16.h>

// ALPHATorch scattering covariance on MI455X (gfx1250).
// All 2D FFTs done as 128x128x128 matmuls with v_wmma_f32_16x16x32_f16
// (f16 inputs, f32 accumulation). DFT matrix is symmetric -> its row-major
// LDS image doubles as the column-major B-operand image, so B fragments are
// contiguous 32B LDS loads. DFT tables are DMA'd into LDS with the Tensor
// Data Mover (tensor_load_to_lds + s_wait_tensorcnt) when available.

typedef _Float16 h16;
typedef __attribute__((ext_vector_type(16))) _Float16 v16h;
typedef __attribute__((ext_vector_type(8)))  float    v8f;

#define PI_F 3.14159265358979323846f
constexpr int   MN      = 16384;     // 128*128
constexpr float INV128  = 1.0f / 128.0f;
constexpr float INV_MN  = 1.0f / 16384.0f;

#if defined(__gfx1250__) && __has_builtin(__builtin_amdgcn_tensor_load_to_lds) && \
    __has_builtin(__builtin_amdgcn_s_wait_tensorcnt)
#define USE_TDM 1
#else
#define USE_TDM 0
#endif

__device__ __forceinline__ v8f wmma_f16(v16h a, v16h b, v8f c) {
  // 8 args: (neg_a, A, neg_b, B, c_mod, C, reuse_a, reuse_b); f16 NEG[1:0] must be 0
  return __builtin_amdgcn_wmma_f32_16x16x32_f16(false, a, false, b, (short)0, c,
                                                false, false);
}

// A fragment: 16x32 f16 tile of row-major [128x128] matrix at (m0, k0).
// Layout: lanes 0-15 row M=lane, halves h -> K=k0+h (matches load_b)
//             lanes 16-31 same rows, K=k0+16+h.
__device__ __forceinline__ v16h load_a(const h16* A, int m0, int k0, int lane) {
  const h16* p = A + (m0 + (lane & 15)) * 128 + k0 + ((lane & 16) ? 16 : 0);
  v16h r;
#pragma unroll
  for (int i = 0; i < 16; ++i) r[i] = p[i];
  return r;
}

// B fragment: 32x16 f16 tile at (k0, n0) of matrix stored COLUMN-major
// (element B[k][n] at Bt[n*128+k]); symmetric matrices can be passed row-major.
// Layout (contiguous k, same as load_a): lanes 0-15 col N=lane, halves h -> K=k0+h; lanes 16-31 K=k0+16+h.
__device__ __forceinline__ v16h load_b(const h16* Bt, int n0, int k0, int lane) {
  const h16* p = Bt + (n0 + (lane & 15)) * 128 + k0 + ((lane & 16) ? 16 : 0);
  v16h r;
#pragma unroll
  for (int i = 0; i < 16; ++i) r[i] = p[i];
  return r;
}

// R = A1*B1 (+ A2*B2 if DUAL) over the full 128x128 output.
// 64 tiles: wave w owns column band n0 = w*16, iterates 8 row tiles (uniform
// trip count -> no EXEC-masked loop around the WMMAs; B fragments of the
// wave's column band are loop-invariant and get hoisted).
template <bool DUAL, class F>
__device__ __forceinline__ void mm2(const h16* A1, const h16* B1,
                                    const h16* A2, const h16* B2, F&& st) {
  const int lane = threadIdx.x & 31;
  const int wid  = (threadIdx.x >> 5) & 7;
  const int n0   = wid << 4;
  for (int tt = 0; tt < 8; ++tt) {
    const int m0 = tt << 4;
    v8f acc = {0.f, 0.f, 0.f, 0.f, 0.f, 0.f, 0.f, 0.f};
#pragma unroll
    for (int k0 = 0; k0 < 128; k0 += 32) {
      if constexpr (DUAL)
        acc = wmma_f16(load_a(A2, m0, k0, lane), load_b(B2, n0, k0, lane), acc);
      acc = wmma_f16(load_a(A1, m0, k0, lane), load_b(B1, n0, k0, lane), acc);
    }
    const int r0 = m0 + ((lane & 16) ? 8 : 0);
    const int c0 = n0 + (lane & 15);
#pragma unroll
    for (int r = 0; r < 8; ++r) st(r0 + r, c0, acc[r]);
  }
}

#if USE_TDM
// One TDM descriptor: copy a contiguous 16384-element f16 plane (32KB) from
// global to LDS. D# per CDNA5 ISA (8.3/8.4): group0 =
// {count=1 | lds_addr | global_addr | type=2}; group1 = {data_size=2B,
// tensor_dim0=16384, tensor_dim1=1, tile_dim0=16384, tile_dim1=1, strides}.
__device__ __forceinline__ void tdm_load_plane(const h16* gsrc, h16* ldst) {
  typedef unsigned int u32x4 __attribute__((ext_vector_type(4)));
  typedef int          i32x4 __attribute__((ext_vector_type(4)));
  typedef int          i32x8 __attribute__((ext_vector_type(8)));
  // generic LDS pointer: low 32 bits are the LDS byte offset
  unsigned loff = (unsigned)(unsigned long long)(const void*)ldst;
  unsigned long long ga = (unsigned long long)gsrc;
  u32x4 g0 = { 1u,                                        // count=1 (valid)
               loff,                                      // lds_addr
               (unsigned)ga,                              // global_addr[31:0]
               ((unsigned)(ga >> 32) & 0x01FFFFFFu) | (2u << 30) }; // [56:32] | type=2
  i32x8 g1 = { (int)(1u << 16),        // workgroup_mask=0, data_size=1 (2 bytes)
               (int)(16384u << 16),    // tensor_dim0[15:0] << 16
               (int)(1u << 16),        // tensor_dim0[31:16]=0 | tensor_dim1[15:0]=1
               (int)(16384u << 16),    // tensor_dim1[31:16]=0 | tile_dim0=16384
               1,                      // tile_dim1=1 | tile_dim2=0
               16384,                  // tensor_dim0_stride[31:0]
               (int)(16384u << 16),    // stride0[47:32]=0 | tensor_dim1_stride[15:0]
               0 };                    // tensor_dim1_stride[47:16]
  i32x4 z4 = {0, 0, 0, 0};
#if __clang_major__ >= 23
  i32x8 z8 = {0, 0, 0, 0, 0, 0, 0, 0};
  __builtin_amdgcn_tensor_load_to_lds(g0, g1, z4, z4, z8, 0);
#else
  __builtin_amdgcn_tensor_load_to_lds(g0, g1, z4, z4, 0);
#endif
}
#endif

// Bring DFT tables into LDS; Sn = -S (chained WMMA can only add, so we
// pre-negate the sine table to realize complex subtractions).
__device__ __forceinline__ void fill_tables(const h16* gF, const h16* gS,
                                            h16* Fr, h16* S, h16* Sn) {
#if USE_TDM
  if (threadIdx.x == 0) {               // TDM ignores EXEC; issue once (wave 0)
    tdm_load_plane(gF, Fr);
    tdm_load_plane(gS, S);
  }
  if (threadIdx.x < 32) __builtin_amdgcn_s_wait_tensorcnt(0);
  __syncthreads();                      // publish DMA'd tables to all waves
  for (int i = threadIdx.x; i < MN; i += blockDim.x)
    Sn[i] = (h16)(-(float)S[i]);
#else
  for (int i = threadIdx.x; i < MN; i += blockDim.x) {
    Fr[i] = gF[i];
    h16 s = gS[i];
    S[i]  = s;
    Sn[i] = (h16)(-(float)s);
  }
#endif
}

// ---------------- kernel 0: DFT twiddle tables (f16) --------------------------
__global__ __launch_bounds__(256) void k_setup(h16* gF, h16* gS) {
  int i = blockIdx.x * 256 + threadIdx.x;       // 16384 total
  int k = i >> 7, n = i & 127;
  float th = (2.0f * PI_F / 128.0f) * (float)((k * n) & 127);
  gF[i] = (h16)__cosf(th);                      // Re(W), forward F = Fr - i*S
  gS[i] = (h16)__sinf(th);
}

// ---------------- kernel 1: hatx = F x F  (one WG per batch) ------------------
__global__ __launch_bounds__(256) void k_fftx(const float* __restrict__ x,
                                              const h16* __restrict__ gF,
                                              const h16* __restrict__ gS,
                                              h16* __restrict__ hatx /*[2][2][MN]*/) {
  __shared__ h16 Fr[MN], S[MN], Sn[MN];
  __shared__ h16 Xt[MN], T1[MN], T2[MN];
  const int b = blockIdx.x;
  fill_tables(gF, gS, Fr, S, Sn);
  const float* xb = x + b * MN;                 // NC == 1
  for (int i = threadIdx.x; i < MN; i += 256) { // stage X transposed (B operand)
    int k = i >> 7, n = i & 127;
    Xt[n * 128 + k] = (h16)xb[i];
  }
  __syncthreads();
  // T = F*X: T1 = Fr*X, T2 = S*X  (so Im(T) = -T2)
  mm2<false>(Fr, Xt, nullptr, nullptr, [&](int r, int c, float v) { T1[r * 128 + c] = (h16)v; });
  mm2<false>(S,  Xt, nullptr, nullptr, [&](int r, int c, float v) { T2[r * 128 + c] = (h16)v; });
  __syncthreads();
  h16* hr = hatx + b * 2 * MN;
  h16* hi = hr + MN;
  // hatx = T*F : Re = T1*Fr - T2*S ; Im = -(T2*Fr + T1*S)
  mm2<true>(T1, Fr, T2, Sn, [&](int r, int c, float v) { hr[r * 128 + c] = (h16)v; });
  mm2<true>(T2, Fr, T1, S,  [&](int r, int c, float v) { hi[r * 128 + c] = (h16)(-v); });
}

// ---------------- kernel 2: per-channel (162 WGs: 2 batches x (80 psi + phi)) -
// Y = hatx .* filt (scaled);  xpsi = relu(Re(F* Y F*));  u = (xpsi-mean)/std/128;
// hu = F u F  -> f16 spectra in workspace.
__global__ __launch_bounds__(256) void k_chan(const h16* __restrict__ gF,
                                              const h16* __restrict__ gS,
                                              const h16* __restrict__ hatx,
                                              const float* __restrict__ hatpsi,
                                              const float* __restrict__ hatphi,
                                              h16* __restrict__ hu /*[2][81][2][MN]*/) {
  __shared__ h16 Fr[MN], S[MN], Sn[MN];
  __shared__ h16 B1[MN], B2[MN], B3[MN], B4[MN];
  __shared__ float xp[MN];
  __shared__ float rs[256], rq[256];
  const int b  = blockIdx.x / 81;
  const int ch = blockIdx.x % 81;
  const h16* hr = hatx + b * 2 * MN;
  const h16* hi = hr + MN;
  // warm caches for this block's spectra while tables DMA into LDS
  __builtin_prefetch(hr + threadIdx.x * 64, 0, 0);   // 256*128B = full 32KB plane
  __builtin_prefetch(hi + threadIdx.x * 64, 0, 0);
  fill_tables(gF, gS, Fr, S, Sn);

  // pointwise spectral filter, staged transposed (B operand); 1/128 keeps f16 in range
  if (ch < 80) {
    const int j = ch >> 4, l = (ch >> 2) & 3, a = ch & 3;
    const float* psi = hatpsi + (j * 4 + l) * MN;
    const float ca = (a == 0) ? 1.f : (a == 2) ? -1.f : 0.f;  // Re(e^{i pi a/2})
    const float sa = (a == 1) ? 1.f : (a == 3) ? -1.f : 0.f;  // Im
    for (int i = threadIdx.x; i < MN; i += 256) {
      float p = psi[i] * INV128;
      float re = (float)hr[i], im = (float)hi[i];
      int k = i >> 7, n = i & 127;
      B1[n * 128 + k] = (h16)((re * ca - im * sa) * p);
      B2[n * 128 + k] = (h16)((re * sa + im * ca) * p);
    }
  } else {
    for (int i = threadIdx.x; i < MN; i += 256) {
      float p = hatphi[i] * INV128;
      int k = i >> 7, n = i & 127;
      B1[n * 128 + k] = (h16)((float)hr[i] * p);
      B2[n * 128 + k] = (h16)((float)hi[i] * p);
    }
  }
  __syncthreads();
  // G = F* Y  (F* = Fr + iS):  Gr = Fr*Yr - S*Yi ; Gi = Fr*Yi + S*Yr
  mm2<true>(Fr, B1, Sn, B2, [&](int r, int c, float v) { B3[r * 128 + c] = (h16)v; });
  mm2<true>(Fr, B2, S,  B1, [&](int r, int c, float v) { B4[r * 128 + c] = (h16)v; });
  __syncthreads();
  // xpsi_pre = Re(G F*) = Gr*Fr - Gi*S ; ReLU only for psi channels
  const bool dorelu = (ch < 80);
  mm2<true>(B3, Fr, B4, Sn, [&](int r, int c, float v) {
    xp[r * 128 + c] = dorelu ? fmaxf(v, 0.f) : v;
  });
  __syncthreads();
  // block reduction: mean & mean-of-squares over the plane
  float s0 = 0.f, s1 = 0.f;
  for (int i = threadIdx.x; i < MN; i += 256) { float v = xp[i]; s0 += v; s1 += v * v; }
  rs[threadIdx.x] = s0; rq[threadIdx.x] = s1;
  __syncthreads();
  for (int o = 128; o > 0; o >>= 1) {
    if (threadIdx.x < (unsigned)o) { rs[threadIdx.x] += rs[threadIdx.x + o]; rq[threadIdx.x] += rq[threadIdx.x + o]; }
    __syncthreads();
  }
  const float mean = rs[0] * INV_MN;
  const float var  = rq[0] * INV_MN - mean * mean;
  const float rstd = rsqrtf(fmaxf(var, 1e-30f)) * INV128;  // fold 1/sqrt(MN)
  // u' staged transposed into B1 (B operand of forward DFT)
  for (int i = threadIdx.x; i < MN; i += 256) {
    int k = i >> 7, n = i & 127;
    B1[n * 128 + k] = (h16)((xp[i] - mean) * rstd);
  }
  __syncthreads();
  // T = F*u': T1=Fr*u'(->B3), T2=S*u'(->B4)
  mm2<false>(Fr, B1, nullptr, nullptr, [&](int r, int c, float v) { B3[r * 128 + c] = (h16)v; });
  mm2<false>(S,  B1, nullptr, nullptr, [&](int r, int c, float v) { B4[r * 128 + c] = (h16)v; });
  __syncthreads();
  h16* hur = hu + (size_t)((b * 81 + ch) * 2) * MN;
  h16* hui = hur + MN;
  mm2<true>(B3, Fr, B4, Sn, [&](int r, int c, float v) { hur[r * 128 + c] = (h16)v; });
  mm2<true>(B4, Fr, B3, S,  [&](int r, int c, float v) { hui[r * 128 + c] = (h16)(-v); });
}

// ---------------- kernel 3: per-pair correlation (1922 WGs) -------------------
// Z' = hu1 .* conj(hu2); corr = Re(F* Z' F*) / MN * mask[shifted]
__global__ __launch_bounds__(256) void k_pair(const h16* __restrict__ gF,
                                              const h16* __restrict__ gS,
                                              const h16* __restrict__ hu,
                                              const int* __restrict__ la1,
                                              const int* __restrict__ la2,
                                              const int* __restrict__ shifted,
                                              const float* __restrict__ masks,
                                              float* __restrict__ out) {
  __shared__ h16 Fr[MN], S[MN], Sn[MN];
  __shared__ h16 B1[MN], B2[MN], B3[MN], B4[MN];
  const int p = blockIdx.x >> 1;   // 0..960 (960 = low-pass phi autocorrelation)
  const int b = blockIdx.x & 1;
  const int c1 = (p < 960) ? la1[p] : 80;
  const int c2 = (p < 960) ? la2[p] : 80;
  const int sh = (p < 960) ? shifted[p] : 5;   // union mask index J
  const h16* ar = hu + (size_t)((b * 81 + c1) * 2) * MN;
  const h16* ai = ar + MN;
  const h16* br = hu + (size_t)((b * 81 + c2) * 2) * MN;
  const h16* bi = br + MN;
  // prefetch the four 32KB spectra planes (128B line per thread covers a plane)
  __builtin_prefetch(ar + threadIdx.x * 64, 0, 0);
  __builtin_prefetch(ai + threadIdx.x * 64, 0, 0);
  __builtin_prefetch(br + threadIdx.x * 64, 0, 0);
  __builtin_prefetch(bi + threadIdx.x * 64, 0, 0);
  fill_tables(gF, gS, Fr, S, Sn);
  for (int i = threadIdx.x; i < MN; i += 256) {
    float x1r = (float)ar[i], x1i = (float)ai[i];
    float x2r = (float)br[i], x2i = (float)bi[i];
    int k = i >> 7, n = i & 127;
    B1[n * 128 + k] = (h16)(x1r * x2r + x1i * x2i);   // Re(F1 conj(F2)) / MN
    B2[n * 128 + k] = (h16)(x1i * x2r - x1r * x2i);   // Im
  }
  __syncthreads();
  mm2<true>(Fr, B1, Sn, B2, [&](int r, int c, float v) { B3[r * 128 + c] = (h16)v; });
  mm2<true>(Fr, B2, S,  B1, [&](int r, int c, float v) { B4[r * 128 + c] = (h16)v; });
  __syncthreads();
  const float* mk = masks + (size_t)sh * MN;
  float* op = out + (size_t)(b * 961 + p) * MN;
  mm2<true>(B3, Fr, B4, Sn, [&](int r, int c, float v) {
    int i = r * 128 + c;
    op[i] = v * INV_MN * mk[i];
  });
}

extern "C" void kernel_launch(void* const* d_in, const int* in_sizes, int n_in,
                              void* d_out, int out_size, void* d_ws, size_t ws_size,
                              hipStream_t stream) {
  (void)in_sizes; (void)n_in; (void)out_size; (void)ws_size;
  const float* x       = (const float*)d_in[0];  // (2,1,128,128)
  const float* hatpsi  = (const float*)d_in[1];  // (5,4,128,128)
  const float* hatphi  = (const float*)d_in[2];  // (128,128)
  const float* masks   = (const float*)d_in[3];  // (6,128,128)
  const int*   la1     = (const int*)d_in[4];    // (960)
  const int*   la2     = (const int*)d_in[5];
  const int*   shifted = (const int*)d_in[6];
  float* out = (float*)d_out;                    // (2, 961*128*128)

  // workspace layout (halfs): Fr | S | hatx[2][2] | hu[2][81][2]  ~= 10.8 MB
  h16* gF   = (h16*)d_ws;
  h16* gS   = gF + MN;
  h16* hatx = gS + MN;
  h16* hu   = hatx + (size_t)4 * MN;

  k_setup<<<64,   256, 0, stream>>>(gF, gS);
  k_fftx <<<2,    256, 0, stream>>>(x, gF, gS, hatx);
  k_chan <<<162,  256, 0, stream>>>(gF, gS, hatx, hatpsi, hatphi, hu);
  k_pair <<<1922, 256, 0, stream>>>(gF, gS, hu, la1, la2, shifted, masks, out);
}
